// GATEdgeNet_Regression_61297773249078
// MI455X (gfx1250) — matrix operand order repaired and checked by reading, not compiled, not run on hardware
//
#include <hip/hip_runtime.h>
#include <hip/hip_bf16.h>

#define NODES   40000
#define EDGES   480000
#define GRAPHS  8
#define EPSV    1e-5f
#define SLOPEV  0.2f

typedef __attribute__((ext_vector_type(16))) __bf16 bf16x16;
typedef __attribute__((ext_vector_type(8)))  __bf16 bf16x8;
typedef __attribute__((ext_vector_type(8)))  float  f32x8;

static inline int cdiv(int a, int b) { return (a + b - 1) / b; }

// ---------------------------------------------------------------- utilities
__global__ void fill_f32(float* __restrict__ p, float v, int n) {
    int i = blockIdx.x * blockDim.x + threadIdx.x;
    if (i < n) p[i] = v;
}

__global__ void cast_bf16(const float* __restrict__ in, __bf16* __restrict__ out, int n) {
    int i = blockIdx.x * blockDim.x + threadIdx.x;
    if (i < n) out[i] = (__bf16)in[i];
}

// W is [K x M] row-major, WT becomes [M x K] row-major (B^T for WMMA B-frag loads)
__global__ void transpose_cast_bf16(const float* __restrict__ W, __bf16* __restrict__ WT,
                                    int K, int M) {
    int i = blockIdx.x * blockDim.x + threadIdx.x;
    if (i < K * M) {
        int k = i / M, m = i - k * M;
        WT[m * K + k] = (__bf16)W[i];
    }
}

__global__ void deg_count(const int* __restrict__ dst, float* __restrict__ deg, int e) {
    int i = blockIdx.x * blockDim.x + threadIdx.x;
    if (i < e) atomicAdd(&deg[dst[i]], 1.0f);
}

__device__ inline void atomicMaxF(float* addr, float v) {
    if (v >= 0.0f) atomicMax((int*)addr, __float_as_int(v));
    else           atomicMin((unsigned int*)addr, __float_as_uint(v));
}

// ---------------------------------------------------------------- WMMA GEMM
// C[N x M] = A[N x K] * B[K x M], with B supplied transposed (BT = [M x K]).
// One wave computes one 16x16 output tile; K-loop of v_wmma_f32_16x16x32_bf16.
__global__ void __launch_bounds__(256)
gemm_bf16_wmma(const __bf16* __restrict__ A, const __bf16* __restrict__ BT,
               float* __restrict__ C, int Nrows, int K, int M) {
    const int wave  = threadIdx.x >> 5;
    const int lane  = threadIdx.x & 31;
    const int tilesN = M >> 4;
    const int tile  = blockIdx.x * 8 + wave;            // wave-uniform
    const int total = (Nrows >> 4) * tilesN;
    if (tile >= total) return;                          // whole wave exits: EXEC all-1 for WMMA
    const int tm = (tile / tilesN) << 4;
    const int tn = (tile % tilesN) << 4;

    const int half = lane >> 4;                         // 0: lanes 0-15, 1: lanes 16-31
    const int l16  = lane & 15;
    // A frag: row = tm+l16; per 32-K block lane holds K in {k0..k0+7, k0+16..k0+23}, k0 = half?8:0
    const __bf16* arow = A  + (size_t)(tm + l16) * K + (half ? 8 : 0);
    // B frag: col = tn+l16; per 32-K block lane holds K in {k0..k0+7, k0+16..k0+23}, k0 = half?8:0
    const __bf16* brow = BT + (size_t)(tn + l16) * K + (half ? 8 : 0);

    f32x8 acc = {};
    for (int k = 0; k < K; k += 32) {
        bf16x8 a_lo = *(const bf16x8*)(arow + k);
        bf16x8 a_hi = *(const bf16x8*)(arow + k + 16);
        bf16x8 b_lo = *(const bf16x8*)(brow + k);
        bf16x8 b_hi = *(const bf16x8*)(brow + k + 16);
        bf16x16 av = __builtin_shufflevector(a_lo, a_hi, 0,1,2,3,4,5,6,7,8,9,10,11,12,13,14,15);
        bf16x16 bv = __builtin_shufflevector(b_lo, b_hi, 0,1,2,3,4,5,6,7,8,9,10,11,12,13,14,15);
        acc = __builtin_amdgcn_wmma_f32_16x16x32_bf16(
                  /*neg_a=*/false, av, /*neg_b=*/false, bv,
                  /*c_mod=*/(short)0, acc, /*reuse_a=*/false, /*reuse_b=*/false);
    }
    // C/D layout: VGPR r -> row tm + r + (half?8:0), col tn + l16
    float* cbase = C + (size_t)(tm + (half ? 8 : 0)) * M + tn + l16;
    #pragma unroll
    for (int r = 0; r < 8; ++r) cbase[(size_t)r * M] = acc[r];
}

// ---------------------------------------------------------------- attention
// asn[n,h] = sum_f H[n,h,f]*attn[h,f] ; adn[n,h] = sum_f H[n,h,f]*attn[h,F+f]
__global__ void node_attn(const float* __restrict__ Hf, const float* __restrict__ attn,
                          float* __restrict__ asn, float* __restrict__ adn,
                          int n, int Hh, int F) {
    int i = blockIdx.x * blockDim.x + threadIdx.x;
    if (i >= n * Hh) return;
    int nn = i / Hh, h = i - nn * Hh;
    const float* hrow = Hf + (size_t)nn * (Hh * F) + h * F;
    const float* as = attn + h * 3 * F;
    const float* ad = as + F;
    float s1 = 0.f, s2 = 0.f;
    for (int f = 0; f < F; ++f) { float hv = hrow[f]; s1 += hv * as[f]; s2 += hv * ad[f]; }
    asn[i] = s1; adn[i] = s2;
}

// cedge[h] = sum_f We[h*F+f] * attn[h, 2F+f]   (EDGE_DIM==1 collapses e-term)
__global__ void edge_cedge(const float* __restrict__ We, const float* __restrict__ attn,
                           float* __restrict__ cedge, int Hh, int F) {
    int t = threadIdx.x;
    if (t < Hh) {
        float c = 0.f;
        const float* ae = attn + t * 3 * F + 2 * F;
        for (int f = 0; f < F; ++f) c += We[t * F + f] * ae[f];
        cedge[t] = c;
    }
}

__global__ void edge_pass_a(const float* __restrict__ asn, const float* __restrict__ adn,
                            const float* __restrict__ ef, const float* __restrict__ cedge,
                            const int* __restrict__ src, const int* __restrict__ dst,
                            float* __restrict__ EA, float* __restrict__ amax,
                            int E_, int Hh) {
    int i = blockIdx.x * blockDim.x + threadIdx.x;
    if (i >= E_ * Hh) return;
    int e = i / Hh, h = i - e * Hh;
    float a = asn[src[e] * Hh + h] + adn[dst[e] * Hh + h] + ef[e] * cedge[h];
    a = a > 0.f ? a : SLOPEV * a;                      // leaky relu
    EA[i] = a;
    atomicMaxF(&amax[dst[e] * Hh + h], a);
}

__global__ void edge_pass_b(float* __restrict__ EA, const float* __restrict__ amax,
                            float* __restrict__ denom, const int* __restrict__ dst,
                            int E_, int Hh) {
    int i = blockIdx.x * blockDim.x + threadIdx.x;
    if (i >= E_ * Hh) return;
    int e = i / Hh, h = i - e * Hh;
    float ea = __expf(EA[i] - amax[dst[e] * Hh + h]);
    EA[i] = ea;
    atomicAdd(&denom[dst[e] * Hh + h], ea);
}

// S[dst, j] += (EA[e,h]/denom[dst,h]) * H[src, j]   over E x (H*F) elements
__global__ void edge_aggregate(const float* __restrict__ EA, const float* __restrict__ denom,
                               const float* __restrict__ Hf,
                               const int* __restrict__ src, const int* __restrict__ dst,
                               float* __restrict__ S, int E_, int Hh, int F) {
    int wf = Hh * F;
    int i = blockIdx.x * blockDim.x + threadIdx.x;
    if (i >= E_ * wf) return;
    int e = i / wf, j = i - e * wf, h = j / F;
    int d = dst[e];
    float alpha = EA[e * Hh + h] / denom[d * Hh + h];
    atomicAdd(&S[(size_t)d * wf + j], alpha * Hf[(size_t)src[e] * wf + j]);
}

__global__ void finalize_bn_relu(const float* __restrict__ S, const float* __restrict__ deg,
                                 const float* __restrict__ gamma, const float* __restrict__ beta,
                                 const float* __restrict__ rmean, const float* __restrict__ rvar,
                                 float* __restrict__ out, int n, int wf) {
    int i = blockIdx.x * blockDim.x + threadIdx.x;
    if (i >= n * wf) return;
    int j = i % wf;
    float s = S[i] / fmaxf(deg[i / wf], 1.0f);
    float sc = gamma[j] * rsqrtf(rvar[j] + EPSV);
    float v = (s - rmean[j]) * sc + beta[j];
    out[i] = fmaxf(v, 0.0f);
}

// ---------------------------------------------------------------- pooling + head
__global__ void pool_kernel(const float* __restrict__ x, const int* __restrict__ batch,
                            float* __restrict__ sums, float* __restrict__ counts,
                            int n, int d) {
    int i = blockIdx.x * blockDim.x + threadIdx.x;
    if (i >= n * d) return;
    int nn = i / d, j = i - nn * d;
    int g = batch[nn];
    atomicAdd(&sums[g * d + j], x[i]);
    if (j == 0) atomicAdd(&counts[g], 1.0f);
}

__global__ void head_kernel(const float* __restrict__ sums, const float* __restrict__ counts,
                            const float* __restrict__ bn_gamma, const float* __restrict__ bn_beta,
                            const float* __restrict__ bn_mean,  const float* __restrict__ bn_var,
                            const float* __restrict__ reg_w, const float* __restrict__ reg_b,
                            const float* __restrict__ cpt_w, const float* __restrict__ cpt_b,
                            float* __restrict__ y_out, float* __restrict__ cpt_out) {
    __shared__ float emb[GRAPHS][128];
    int j = threadIdx.x;                                // 0..127
    float sc = bn_gamma[j] * rsqrtf(bn_var[j] + EPSV);
    for (int g = 0; g < GRAPHS; ++g) {
        float v = sums[g * 128 + j] / fmaxf(counts[g], 1.0f);
        emb[g][j] = (v - bn_mean[j]) * sc + bn_beta[j];
    }
    __syncthreads();
    if (j < GRAPHS) {
        float acc = reg_b[0];
        for (int k = 0; k < 128; ++k) acc += emb[j][k] * reg_w[k];
        y_out[j] = acc;
    }
    {
        int g = j >> 4, c = j & 15;                     // 8*16 = 128 concept outputs
        float acc = cpt_b[c];
        for (int k = 0; k < 128; ++k) acc += emb[g][k] * cpt_w[k * 16 + c];
        cpt_out[g * 16 + c] = acc;
    }
}

// ---------------------------------------------------------------- driver
extern "C" void kernel_launch(void* const* d_in, const int* in_sizes, int n_in,
                              void* d_out, int out_size, void* d_ws, size_t ws_size,
                              hipStream_t stream) {
    (void)in_sizes; (void)n_in; (void)out_size; (void)ws_size;
    // pytree order: x, edge_feats, {params leaves sorted: bn_beta,bn_gamma,bn_mean,bn_var,
    // cpt_b,cpt_w, layers[0..2]{We,Wn,attn,beta,gamma,rmean,rvar}, reg_b,reg_w}, src, dst, batch
    const float* x0       = (const float*)d_in[0];
    const float* ef       = (const float*)d_in[1];
    const float* bn_beta  = (const float*)d_in[2];
    const float* bn_gamma = (const float*)d_in[3];
    const float* bn_mean  = (const float*)d_in[4];
    const float* bn_var   = (const float*)d_in[5];
    const float* cpt_b    = (const float*)d_in[6];
    const float* cpt_w    = (const float*)d_in[7];
    const float* reg_b    = (const float*)d_in[29];
    const float* reg_w    = (const float*)d_in[30];
    const int*   src      = (const int*)d_in[31];
    const int*   dst      = (const int*)d_in[32];
    const int*   batch    = (const int*)d_in[33];

    char* ws = (char*)d_ws; size_t off = 0;
    auto carve = [&](size_t bytes) -> void* {
        void* p = ws + off; off += (bytes + 255) & ~(size_t)255; return p;
    };
    float*  X     = (float*)carve((size_t)NODES * 256 * 4);   // layer ping buffer
    float*  Hf    = (float*)carve((size_t)NODES * 256 * 4);   // projected features
    float*  S     = (float*)carve((size_t)NODES * 256 * 4);   // scatter accumulator
    __bf16* Xbf   = (__bf16*)carve((size_t)NODES * 256 * 2);
    __bf16* WT    = (__bf16*)carve((size_t)256 * 256 * 2);
    float*  EA    = (float*)carve((size_t)EDGES * 4 * 4);
    float*  asn   = (float*)carve((size_t)NODES * 4 * 4);
    float*  adn   = (float*)carve((size_t)NODES * 4 * 4);
    float*  amax  = (float*)carve((size_t)NODES * 4 * 4);
    float*  denom = (float*)carve((size_t)NODES * 4 * 4);
    float*  deg   = (float*)carve((size_t)NODES * 4);
    float*  cedge = (float*)carve(256);
    float*  sums  = (float*)carve((size_t)GRAPHS * 128 * 4);
    float*  counts= (float*)carve((size_t)GRAPHS * 4);

    const int B = 256;

    fill_f32<<<cdiv(NODES, B), B, 0, stream>>>(deg, 0.f, NODES);
    deg_count<<<cdiv(EDGES, B), B, 0, stream>>>(dst, deg, EDGES);

    const int in_fs[3]  = {128, 256, 256};
    const int out_fs[3] = {64, 64, 128};
    const int headsL[3] = {4, 4, 1};

    const float* Xin = x0;
    for (int l = 0; l < 3; ++l) {
        const float* We    = (const float*)d_in[8 + 7 * l + 0];
        const float* Wn    = (const float*)d_in[8 + 7 * l + 1];
        const float* attn  = (const float*)d_in[8 + 7 * l + 2];
        const float* beta  = (const float*)d_in[8 + 7 * l + 3];
        const float* gamma = (const float*)d_in[8 + 7 * l + 4];
        const float* rmean = (const float*)d_in[8 + 7 * l + 5];
        const float* rvar  = (const float*)d_in[8 + 7 * l + 6];
        const int K = in_fs[l], F = out_fs[l], H = headsL[l], WF = F * H;

        cast_bf16<<<cdiv(NODES * K, B), B, 0, stream>>>(Xin, Xbf, NODES * K);
        transpose_cast_bf16<<<cdiv(K * WF, B), B, 0, stream>>>(Wn, WT, K, WF);
        {
            int tiles = (NODES / 16) * (WF / 16);
            gemm_bf16_wmma<<<cdiv(tiles, 8), 256, 0, stream>>>(Xbf, WT, Hf, NODES, K, WF);
        }
        fill_f32<<<cdiv(NODES * H, B), B, 0, stream>>>(amax, -3.4e38f, NODES * H);
        fill_f32<<<cdiv(NODES * H, B), B, 0, stream>>>(denom, 0.f, NODES * H);
        fill_f32<<<cdiv(NODES * WF, B), B, 0, stream>>>(S, 0.f, NODES * WF);
        node_attn<<<cdiv(NODES * H, B), B, 0, stream>>>(Hf, attn, asn, adn, NODES, H, F);
        edge_cedge<<<1, 32, 0, stream>>>(We, attn, cedge, H, F);
        edge_pass_a<<<cdiv(EDGES * H, B), B, 0, stream>>>(asn, adn, ef, cedge, src, dst, EA, amax, EDGES, H);
        edge_pass_b<<<cdiv(EDGES * H, B), B, 0, stream>>>(EA, amax, denom, dst, EDGES, H);
        edge_aggregate<<<cdiv(EDGES * WF, B), B, 0, stream>>>(EA, denom, Hf, src, dst, S, EDGES, H, F);

        float* dest = (l < 2) ? X : (float*)d_out;     // layer-3 x goes straight to output
        finalize_bn_relu<<<cdiv(NODES * WF, B), B, 0, stream>>>(S, deg, gamma, beta, rmean, rvar,
                                                                dest, NODES, WF);
        Xin = dest;
    }

    fill_f32<<<cdiv(GRAPHS * 128, B), B, 0, stream>>>(sums, 0.f, GRAPHS * 128);
    fill_f32<<<1, B, 0, stream>>>(counts, 0.f, GRAPHS);
    pool_kernel<<<cdiv(NODES * 128, B), B, 0, stream>>>((const float*)d_out, batch, sums, counts,
                                                        NODES, 128);
    float* y_out   = (float*)d_out + (size_t)NODES * 128;
    float* cpt_out = y_out + GRAPHS;                    // y is [G,1] -> 8 floats
    head_kernel<<<1, 128, 0, stream>>>(sums, counts, bn_gamma, bn_beta, bn_mean, bn_var,
                                       reg_w, reg_b, cpt_w, cpt_b, y_out, cpt_out);
}
